// GlobalGraphConv_30502857736586
// MI455X (gfx1250) — matrix operand order repaired and checked by reading, not compiled, not run on hardware
//
#include <hip/hip_runtime.h>

typedef __attribute__((ext_vector_type(16))) _Float16 v16h;
typedef __attribute__((ext_vector_type(8)))  float    v8f;

#define Bsz 4
#define Cch 64
#define Nn  4096
#define ICh 32

__device__ __forceinline__ float lrelu(float x) { return x > 0.f ? x : 0.2f * x; }

// ---------------------------------------------------------------------------
// Pass 1: per-node projections.
//   gT[b,o,n]  = f16( g_w[o,:] . x[b,:,n] + g_b[o] )      (j-contiguous for WMMA B)
//   a[b,n]     = (theta_w^T cp_w[:IC]) . x[b,:,n] + cp_w[:IC].theta_b
//   p[b,n]     = (phi_w^T   cp_w[IC:]) . x[b,:,n] + cp_w[IC:].phi_b
// ---------------------------------------------------------------------------
__global__ void __launch_bounds__(256) ggc_pass1(
    const float* __restrict__ x, const float* __restrict__ g_w,
    const float* __restrict__ g_b, const float* __restrict__ theta_w,
    const float* __restrict__ theta_b, const float* __restrict__ phi_w,
    const float* __restrict__ phi_b, const float* __restrict__ cp_w,
    _Float16* __restrict__ gT, float* __restrict__ abuf, float* __restrict__ pbuf)
{
    __shared__ float wa[Cch], wp[Cch], beta2[2];
    const int t = threadIdx.x;
    if (t < Cch) {
        float s = 0.f;
        #pragma unroll
        for (int k = 0; k < ICh; ++k) s += cp_w[k] * theta_w[k * Cch + t];
        wa[t] = s;
    } else if (t < 2 * Cch) {
        const int c = t - Cch;
        float s = 0.f;
        #pragma unroll
        for (int k = 0; k < ICh; ++k) s += cp_w[ICh + k] * phi_w[k * Cch + c];
        wp[c] = s;
    } else if (t == 2 * Cch) {
        float s0 = 0.f, s1 = 0.f;
        #pragma unroll
        for (int k = 0; k < ICh; ++k) {
            s0 += cp_w[k] * theta_b[k];
            s1 += cp_w[ICh + k] * phi_b[k];
        }
        beta2[0] = s0; beta2[1] = s1;
    }
    __syncthreads();

    const int b = blockIdx.y;
    const int n = blockIdx.x * blockDim.x + threadIdx.x;

    float xv[Cch];
    #pragma unroll 8
    for (int c = 0; c < Cch; ++c) xv[c] = x[(size_t)(b * Cch + c) * Nn + n];

    float av = beta2[0], pv = beta2[1];
    #pragma unroll 8
    for (int c = 0; c < Cch; ++c) { av += wa[c] * xv[c]; pv += wp[c] * xv[c]; }
    abuf[b * Nn + n] = av;
    pbuf[b * Nn + n] = pv;

    for (int o = 0; o < Cch; ++o) {
        float acc = g_b[o];
        #pragma unroll
        for (int c = 0; c < Cch; ++c) acc += g_w[o * Cch + c] * xv[c];
        gT[(size_t)(b * Cch + o) * Nn + n] = (_Float16)acc;
    }
}

// ---------------------------------------------------------------------------
// Pass 2: pmax[b] = max_j p[b,j]  (one 1024-thread block per batch)
// ---------------------------------------------------------------------------
__global__ void __launch_bounds__(1024) ggc_pass2(
    const float* __restrict__ pbuf, float* __restrict__ pmax)
{
    __shared__ float red[1024];
    const int b = blockIdx.x;
    float m = -3.0e38f;
    for (int j = threadIdx.x; j < Nn; j += 1024) m = fmaxf(m, pbuf[b * Nn + j]);
    red[threadIdx.x] = m;
    __syncthreads();
    #pragma unroll
    for (int s = 512; s > 0; s >>= 1) {
        if (threadIdx.x < s) red[threadIdx.x] = fmaxf(red[threadIdx.x], red[threadIdx.x + s]);
        __syncthreads();
    }
    if (threadIdx.x == 0) pmax[b] = red[0];
}

// ---------------------------------------------------------------------------
// Pass 3: softmax denominators. Row max is lrelu(a_i + pmax) (lrelu monotone).
// One wave32 per row; reduce via __shfl_xor.
// ---------------------------------------------------------------------------
__global__ void __launch_bounds__(256) ggc_pass3(
    const float* __restrict__ abuf, const float* __restrict__ pbuf,
    const float* __restrict__ pmax, float* __restrict__ mbuf,
    float* __restrict__ invS)
{
    const int row  = (blockIdx.x * blockDim.x + threadIdx.x) >> 5; // 0 .. B*Nn-1
    const int lane = threadIdx.x & 31;
    const int b    = row >> 12;                                    // Nn = 4096
    const float ai = abuf[row];
    const float mi = lrelu(ai + pmax[b]);
    const float* pr = pbuf + b * Nn;
    float s = 0.f;
    for (int j = lane; j < Nn; j += 32)
        s += __expf(lrelu(ai + pr[j]) - mi);
    #pragma unroll
    for (int off = 16; off > 0; off >>= 1) s += __shfl_xor(s, off, 32);
    if (lane == 0) { mbuf[row] = mi; invS[row] = 1.f / s; }
}

// ---------------------------------------------------------------------------
// Pass 4: fused  y[b,o,i] = sum_j ( P_ij/S_i + C_k[i,j] ) * g_x[b,j,o]
// One wave per 16-row i-tile. Two-stage software pipeline: stage k+1's
// p / C_k / B-tile loads are issued before stage k's WMMAs, hiding L2
// latency behind the A-tile build (16 exp) + 4 WMMAs of the prior stage.
// Uniform control flow: EXEC all-ones for v_wmma as the ISA requires.
// ---------------------------------------------------------------------------
struct Stage {
    float4 pa, pb, pc, pd;   // p[jb0+0..3], p[jb0+4..7], p[jb1+0..3], p[jb1+4..7]
    float4 ca, cb, cc, cd;   // C_k row, same element mapping as p
    v16h   B0, B1, B2, B3;   // gT tiles for o-blocks 0..3
};

__device__ __forceinline__ void load_stage(
    Stage& s, const float* __restrict__ prow, const float* __restrict__ crow,
    const v16h* __restrict__ gp0, const v16h* __restrict__ gp1,
    const v16h* __restrict__ gp2, const v16h* __restrict__ gp3,
    int j0, int h)
{
    const int jb0 = j0 + 16 * h;
    const int jb1 = jb0 + 8;
    s.pa = *(const float4*)(prow + jb0);
    s.pb = *(const float4*)(prow + jb0 + 4);
    s.pc = *(const float4*)(prow + jb1);
    s.pd = *(const float4*)(prow + jb1 + 4);
    s.ca = *(const float4*)(crow + jb0);
    s.cb = *(const float4*)(crow + jb0 + 4);
    s.cc = *(const float4*)(crow + jb1);
    s.cd = *(const float4*)(crow + jb1 + 4);
    const int ji = j0 >> 4;                 // v16h units (32 B)
    s.B0 = gp0[ji];
    s.B1 = gp1[ji];
    s.B2 = gp2[ji];
    s.B3 = gp3[ji];
}

__device__ __forceinline__ v16h build_A(const Stage& s, float ai, float mi, float isv)
{
    const float pv[16] = { s.pa.x, s.pa.y, s.pa.z, s.pa.w,
                           s.pb.x, s.pb.y, s.pb.z, s.pb.w,
                           s.pc.x, s.pc.y, s.pc.z, s.pc.w,
                           s.pd.x, s.pd.y, s.pd.z, s.pd.w };
    const float cv[16] = { s.ca.x, s.ca.y, s.ca.z, s.ca.w,
                           s.cb.x, s.cb.y, s.cb.z, s.cb.w,
                           s.cc.x, s.cc.y, s.cc.z, s.cc.w,
                           s.cd.x, s.cd.y, s.cd.z, s.cd.w };
    v16h A;
    #pragma unroll
    for (int e = 0; e < 16; ++e)
        A[e] = (_Float16)(__expf(lrelu(ai + pv[e]) - mi) * isv + cv[e]);
    return A;
}

__global__ void __launch_bounds__(256) ggc_pass4(
    const float* __restrict__ abuf, const float* __restrict__ pbuf,
    const float* __restrict__ mbuf, const float* __restrict__ invSbuf,
    const float* __restrict__ Ck, const _Float16* __restrict__ gT,
    float* __restrict__ out)
{
    const int wave = threadIdx.x >> 5;
    const int lane = threadIdx.x & 31;
    const int tile = blockIdx.x * 8 + wave;   // B * (Nn/16) = 1024 tiles
    const int b    = tile >> 8;               // 256 i-tiles per batch
    const int i0   = (tile & 255) << 4;

    const int mrow = lane & 15;
    const int h    = lane >> 4;
    const int i    = i0 + mrow;

    const float ai  = abuf[b * Nn + i];
    const float mi  = mbuf[b * Nn + i];
    const float isv = invSbuf[b * Nn + i];
    const float* prow = pbuf + b * Nn;
    const float* crow = Ck + (size_t)i * Nn;

    // B operand: lane -> column o = oc*16 + mrow, 16 consecutive j halfs at +h*16
    const v16h* gp0 = (const v16h*)(gT + (size_t)(b * Cch +  0 + mrow) * Nn + h * 16);
    const v16h* gp1 = (const v16h*)(gT + (size_t)(b * Cch + 16 + mrow) * Nn + h * 16);
    const v16h* gp2 = (const v16h*)(gT + (size_t)(b * Cch + 32 + mrow) * Nn + h * 16);
    const v16h* gp3 = (const v16h*)(gT + (size_t)(b * Cch + 48 + mrow) * Nn + h * 16);

    v8f acc0 = {}, acc1 = {}, acc2 = {}, acc3 = {};

    Stage s0, s1;
    load_stage(s0, prow, crow, gp0, gp1, gp2, gp3, 0, h);

    for (int j0 = 0; j0 < Nn; j0 += 64) {
        // prefetch odd stage while even stage computes
        load_stage(s1, prow, crow, gp0, gp1, gp2, gp3, j0 + 32, h);
        {
            v16h A = build_A(s0, ai, mi, isv);
            acc0 = __builtin_amdgcn_wmma_f32_16x16x32_f16(false, A, false, s0.B0, (short)0, acc0, false, false);
            acc1 = __builtin_amdgcn_wmma_f32_16x16x32_f16(false, A, false, s0.B1, (short)0, acc1, false, false);
            acc2 = __builtin_amdgcn_wmma_f32_16x16x32_f16(false, A, false, s0.B2, (short)0, acc2, false, false);
            acc3 = __builtin_amdgcn_wmma_f32_16x16x32_f16(false, A, false, s0.B3, (short)0, acc3, false, false);
        }
        // prefetch next even stage while odd stage computes (guarded: uniform branch)
        if (j0 + 64 < Nn)
            load_stage(s0, prow, crow, gp0, gp1, gp2, gp3, j0 + 64, h);
        {
            v16h A = build_A(s1, ai, mi, isv);
            acc0 = __builtin_amdgcn_wmma_f32_16x16x32_f16(false, A, false, s1.B0, (short)0, acc0, false, false);
            acc1 = __builtin_amdgcn_wmma_f32_16x16x32_f16(false, A, false, s1.B1, (short)0, acc1, false, false);
            acc2 = __builtin_amdgcn_wmma_f32_16x16x32_f16(false, A, false, s1.B2, (short)0, acc2, false, false);
            acc3 = __builtin_amdgcn_wmma_f32_16x16x32_f16(false, A, false, s1.B3, (short)0, acc3, false, false);
        }
    }

    // D layout: lane column n = mrow -> o = oc*16 + mrow; VGPR v -> row M = v + 8h.
    // Rows are contiguous in i (innermost N dim of out (B,C,N)).
    v8f accs[4] = {acc0, acc1, acc2, acc3};
    #pragma unroll
    for (int oc = 0; oc < 4; ++oc) {
        float* op = out + (size_t)(b * Cch + oc * 16 + mrow) * Nn + i0 + 8 * h;
        #pragma unroll
        for (int v = 0; v < 8; ++v) op[v] = accs[oc][v];
    }
}

// ---------------------------------------------------------------------------
extern "C" void kernel_launch(void* const* d_in, const int* in_sizes, int n_in,
                              void* d_out, int out_size, void* d_ws, size_t ws_size,
                              hipStream_t stream)
{
    const float* x       = (const float*)d_in[0];
    const float* g_w     = (const float*)d_in[1];
    const float* g_b     = (const float*)d_in[2];
    const float* theta_w = (const float*)d_in[3];
    const float* theta_b = (const float*)d_in[4];
    const float* phi_w   = (const float*)d_in[5];
    const float* phi_b   = (const float*)d_in[6];
    const float* cp_w    = (const float*)d_in[7];
    const float* Ck      = (const float*)d_in[8];
    float* out = (float*)d_out;

    char* ws = (char*)d_ws;
    const size_t gT_bytes = (size_t)Bsz * Cch * Nn * sizeof(_Float16); // 2 MB
    const size_t bn_bytes = (size_t)Bsz * Nn * sizeof(float);          // 64 KB
    _Float16* gT  = (_Float16*)ws;
    float* abuf   = (float*)(ws + gT_bytes);
    float* pbuf   = (float*)(ws + gT_bytes + 1 * bn_bytes);
    float* mbuf   = (float*)(ws + gT_bytes + 2 * bn_bytes);
    float* invS   = (float*)(ws + gT_bytes + 3 * bn_bytes);
    float* pmax   = (float*)(ws + gT_bytes + 4 * bn_bytes);

    ggc_pass1<<<dim3(Nn / 256, Bsz), 256, 0, stream>>>(
        x, g_w, g_b, theta_w, theta_b, phi_w, phi_b, cp_w, gT, abuf, pbuf);
    ggc_pass2<<<Bsz, 1024, 0, stream>>>(pbuf, pmax);
    ggc_pass3<<<(Bsz * Nn) / 8, 256, 0, stream>>>(abuf, pbuf, pmax, mbuf, invS);
    ggc_pass4<<<(Bsz * (Nn / 16)) / 8, 256, 0, stream>>>(
        abuf, pbuf, mbuf, invS, Ck, gT, out);
}
